// VSSBlock_90984587198520
// MI455X (gfx1250) — matrix operand order repaired and checked by reading, not compiled, not run on hardware
//
#include <hip/hip_runtime.h>

// ---------------------------------------------------------------------------
// ChannelMamba (bimamba v2) for MI455X / gfx1250.
// b=4, d_inner=256, l=4096, n=16, r=16.
// GEMMs: bf16 WMMA 16x16x32 (f32 accumulate). Scan: 3-phase chunked parallel
// scan (64 chunks x 64 steps) with ASYNC global->LDS staging of B/C rows.
// All intermediates L2-resident (192 MB L2; working set ~175 MB ws + 34 MB IO).
// ---------------------------------------------------------------------------

typedef __bf16 bf16;
typedef __attribute__((ext_vector_type(16))) bf16  v16bf;
typedef __attribute__((ext_vector_type(8)))  float v8f;

#define B_   4
#define DI_  256
#define L_   4096
#define N_   16
#define R_   16
#define E2_  512          // 2*d_inner
#define CH_  64           // scan chunks
#define CS_  64           // chunk size (CH_*CS_ == L_)

// ---------------------------------------------------------------------------
// CDNA5 async memory->LDS staging (ASYNCcnt path, ISA §15.18.3 op 98).
// Per-lane: 16 bytes global -> 16 bytes LDS. LDS offset = flat addr[31:0].
// ---------------------------------------------------------------------------
__device__ __forceinline__ void async_ld_lds_b128(const float* gsrc, const float* lds_dst)
{
    unsigned long long ga = (unsigned long long)(const void*)gsrc;
    unsigned loff = (unsigned)(unsigned long long)(const void*)lds_dst;
    asm volatile("global_load_async_to_lds_b128 %0, %1, off"
                 :: "v"(loff), "v"(ga) : "memory");
}
__device__ __forceinline__ void wait_async0()
{
    asm volatile("s_wait_asynccnt 0x0" ::: "memory");
}

// ---------------------------------------------------------------------------
// WMMA helpers. A: MxK row-major bf16 (tile row 0 at pointer). BT: NxK
// row-major bf16 (activation stored l-major, d contiguous == B transposed).
// Fragment layout per CDNA5 ISA 7.12.2 (16-bit A 16x32): lanes 0-15 hold
// K {0..7,16..23}, lanes 16-31 hold K {8..15,24..31}; B uses the same
// K order per lane half.
// ---------------------------------------------------------------------------
__device__ __forceinline__ v8f wmma_gemm_tile(const bf16* __restrict__ A,
                                              const bf16* __restrict__ BT,
                                              int K, int lda, int ldb, v8f acc)
{
    const int lane = threadIdx.x & 31;
    const int half = lane >> 4;
    const int mr   = lane & 15;
    for (int k0 = 0; k0 < K; k0 += 32) {
        v16bf av, bv;
        const bf16* pa0 = A  + mr * lda + k0 + half * 8;
        const bf16* pa1 = pa0 + 16;
        const bf16* pb0 = BT + mr * ldb + k0 + half * 8;
        const bf16* pb1 = pb0 + 16;
#pragma unroll
        for (int i = 0; i < 8; ++i) { av[i] = pa0[i]; av[i + 8] = pa1[i]; }
#pragma unroll
        for (int i = 0; i < 8; ++i) { bv[i] = pb0[i]; bv[i + 8] = pb1[i]; }
        acc = __builtin_amdgcn_wmma_f32_16x16x32_bf16(
            /*neg_a=*/false, av, /*neg_b=*/false, bv,
            /*c_mod=*/(short)0, acc, /*reuse_a=*/false, /*reuse_b=*/false);
    }
    return acc;
}

__device__ __forceinline__ void wmma_store_tile(float* __restrict__ C, int ldc, v8f acc)
{
    const int lane = threadIdx.x & 31;
    const int half = lane >> 4;
    const int mr   = lane & 15;
#pragma unroll
    for (int r = 0; r < 8; ++r)
        C[(size_t)(r + 8 * half) * ldc + mr] = acc[r];
}

// ---------------------------------------------------------------------------
// Prep: casts / transposes
// ---------------------------------------------------------------------------
__global__ void k_cast(const float* __restrict__ s, bf16* __restrict__ d, int n)
{
    int i = blockIdx.x * blockDim.x + threadIdx.x;
    if (i < n) d[i] = (bf16)s[i];
}

// u (b,d,l) f32 -> ubf (b,l,d) bf16
__global__ void k_castu(const float* __restrict__ u, bf16* __restrict__ ubf)
{
    size_t idx = (size_t)blockIdx.x * blockDim.x + threadIdx.x;   // b:2 d:8 l:12
    int l = idx & 4095, d = (idx >> 12) & 255, b = idx >> 20;
    ubf[((size_t)b * L_ + l) * DI_ + d] = (bf16)u[idx];
}

// ---------------------------------------------------------------------------
// in_proj: xz[b,e,l] = sum_d W[e,d]*u[b,d,l]   (M=512,N=4096,K=256)
// ---------------------------------------------------------------------------
__global__ void k_inproj(const bf16* __restrict__ W, const bf16* __restrict__ ubf,
                         float* __restrict__ xz)
{
    int lt = blockIdx.x, et = blockIdx.y, b = blockIdx.z;
    v8f acc = {};
    const bf16* A  = W + (size_t)et * 16 * DI_;
    const bf16* BT = ubf + ((size_t)b * L_ + (size_t)lt * 16) * DI_;
    acc = wmma_gemm_tile(A, BT, DI_, DI_, DI_, acc);
    float* C = xz + ((size_t)b * E2_ + (size_t)et * 16) * L_ + (size_t)lt * 16;
    wmma_store_tile(C, L_, acc);
}

// ---------------------------------------------------------------------------
// Depthwise conv: spatial dim 1 x d_inner, pad 1 -> only middle kernel row.
// x[b,d,l] = cw[l,1,0]*x[d-1] + cw[l,1,1]*x[d] + cw[l,1,2]*x[d+1] + cb[l]
// Writes xT (b,l,d) f32, xbf (b,l,d) bf16, zT (b,l,d) f32.
// ---------------------------------------------------------------------------
__global__ void k_conv(const float* __restrict__ xz, const float* __restrict__ cw,
                       const float* __restrict__ cb,
                       float* __restrict__ xT, bf16* __restrict__ xbf,
                       float* __restrict__ zT)
{
    size_t idx = (size_t)blockIdx.x * blockDim.x + threadIdx.x;
    int l = idx & 4095, d = (idx >> 12) & 255, b = idx >> 20;
    const float* xr = xz + ((size_t)b * E2_ + d) * L_ + l;
    float xm = (d > 0)   ? xr[-(int)L_] : 0.f;
    float x0 = xr[0];
    float xp = (d < 255) ? xr[L_] : 0.f;
    const float* w = cw + (size_t)l * 9 + 3;          // middle kernel row
    float v = w[0] * xm + w[1] * x0 + w[2] * xp + cb[l];
    size_t to = ((size_t)b * L_ + l) * DI_ + d;
    xT[to]  = v;
    xbf[to] = (bf16)v;
    zT[to]  = xz[((size_t)b * E2_ + DI_ + d) * L_ + l];
}

// ---------------------------------------------------------------------------
// x_proj (both branches): dbl[b,k,l] = sum_d XW[k,d]*x[b,d,l]  (M=48,K=256)
// ---------------------------------------------------------------------------
__global__ void k_xproj(const bf16* __restrict__ xw, const bf16* __restrict__ xwb,
                        const bf16* __restrict__ xbf,
                        float* __restrict__ dblf, float* __restrict__ dblb)
{
    int lt = blockIdx.x, mt = blockIdx.y;
    int z = blockIdx.z, b = z >> 1, br = z & 1;
    v8f acc = {};
    const bf16* A  = (br ? xwb : xw) + (size_t)mt * 16 * DI_;
    const bf16* BT = xbf + ((size_t)b * L_ + (size_t)lt * 16) * DI_;
    acc = wmma_gemm_tile(A, BT, DI_, DI_, DI_, acc);
    float* C = (br ? dblb : dblf) + ((size_t)b * 48 + (size_t)mt * 16) * L_ + (size_t)lt * 16;
    wmma_store_tile(C, L_, acc);
}

// ---------------------------------------------------------------------------
// dt_proj + softplus(dt + dt_bias): K=16, pure f32 VALU (tiny GEMM).
// deltaT layout (b*2+br, l, d) f32 for coalesced scan reads.
// ---------------------------------------------------------------------------
__global__ void k_dt(const float* __restrict__ dblf, const float* __restrict__ dblb,
                     const float* __restrict__ dtw, const float* __restrict__ dtwb,
                     const float* __restrict__ dt_bias, float* __restrict__ deltaT)
{
    size_t idx = (size_t)blockIdx.x * blockDim.x + threadIdx.x;   // br:1 b:2 d:8 l:12
    int l = idx & 4095, d = (idx >> 12) & 255, b = (idx >> 20) & 3, br = idx >> 22;
    const float* dbl = br ? dblb : dblf;
    const float* w   = br ? dtwb : dtw;
    float acc = dt_bias[d];
#pragma unroll
    for (int r = 0; r < R_; ++r)
        acc += w[d * R_ + r] * dbl[((size_t)b * 48 + r) * L_ + l];
    float sp = (acc > 15.f) ? acc : log1pf(__expf(acc));          // softplus
    deltaT[(((size_t)(b * 2 + br)) * L_ + l) * DI_ + d] = sp;
}

// ---------------------------------------------------------------------------
// Scan phase 1: per-chunk local recurrence (h0=0). Emits per-(chunk,d):
//   hfinal[.,n] and dsum = sum(delta) (chunk multiplier == exp(A*dsum)).
// B rows staged to LDS via global_load_async_to_lds_b128 (ASYNCcnt).
// ---------------------------------------------------------------------------
__global__ void k_scan_local(const float* __restrict__ deltaT, const float* __restrict__ xT,
                             const float* __restrict__ dblf, const float* __restrict__ dblb,
                             const float* __restrict__ A_log, const float* __restrict__ A_b_log,
                             float* __restrict__ hfinal, float* __restrict__ dsum)
{
    __shared__ float sB[N_][CS_];
    int chunk = blockIdx.x, b = blockIdx.y, br = blockIdx.z;
    int d = threadIdx.x;
    int bbr = b * 2 + br;
    const float* dbl = br ? dblb : dblf;
    {   // 256 lanes x 16B = whole 4KB tile in one async op per wave
        int n = threadIdx.x >> 4, t4 = (threadIdx.x & 15) * 4;
        async_ld_lds_b128(dbl + ((size_t)b * 48 + R_ + n) * L_ + chunk * CS_ + t4,
                          &sB[n][t4]);
        wait_async0();
    }
    __syncthreads();
    const float* Alog = br ? A_b_log : A_log;
    float Adn[N_];
#pragma unroll
    for (int n = 0; n < N_; ++n) Adn[n] = -__expf(Alog[d * N_ + n]);
    float h[N_] = {0.f};
    float ds = 0.f;
    const float* dT = deltaT + (((size_t)bbr) * L_ + (size_t)chunk * CS_) * DI_ + d;
    for (int tt = 0; tt < CS_; ++tt) {
        int t  = chunk * CS_ + tt;
        int tu = br ? (L_ - 1 - t) : t;
        float delta = dT[(size_t)tt * DI_];
        float u = xT[((size_t)b * L_ + tu) * DI_ + d];
        float du = delta * u;
        ds += delta;
#pragma unroll
        for (int n = 0; n < N_; ++n)
            h[n] = h[n] * __expf(delta * Adn[n]) + du * sB[n][tt];
    }
    size_t base = (((size_t)bbr * CH_ + chunk) * DI_ + d);
    dsum[base] = ds;
#pragma unroll
    for (int n = 0; n < N_; ++n) hfinal[base * N_ + n] = h[n];
}

// ---------------------------------------------------------------------------
// Scan phase 2: carry scan across 64 chunks, one thread per (b,br,d,n).
// ---------------------------------------------------------------------------
__global__ void k_scan_carry(const float* __restrict__ hfinal, const float* __restrict__ dsum,
                             const float* __restrict__ A_log, const float* __restrict__ A_b_log,
                             float* __restrict__ hin)
{
    int idx = blockIdx.x * blockDim.x + threadIdx.x;      // bbr:3 d:8 n:4
    if (idx >= B_ * 2 * DI_ * N_) return;
    int n = idx & 15, d = (idx >> 4) & 255, bbr = idx >> 12;
    const float* Alog = (bbr & 1) ? A_b_log : A_log;
    float Adn = -__expf(Alog[d * N_ + n]);
    float h = 0.f;
    for (int c = 0; c < CH_; ++c) {
        size_t base = (((size_t)bbr * CH_ + c) * DI_ + d);
        hin[base * N_ + n] = h;
        h = h * __expf(Adn * dsum[base]) + hfinal[base * N_ + n];
    }
}

// ---------------------------------------------------------------------------
// Scan phase 3: re-scan each chunk from its carried-in state, emit
//   y[t] = (sum_n h*C + D*u) * silu(z[t])   into ybuf (bbr,d,l).
// B and C rows staged to LDS via async copies.
// ---------------------------------------------------------------------------
__global__ void k_scan_emit(const float* __restrict__ deltaT, const float* __restrict__ xT,
                            const float* __restrict__ zT,
                            const float* __restrict__ dblf, const float* __restrict__ dblb,
                            const float* __restrict__ A_log, const float* __restrict__ A_b_log,
                            const float* __restrict__ Dv, const float* __restrict__ Dbv,
                            const float* __restrict__ hin, float* __restrict__ ybuf)
{
    __shared__ float sB[N_][CS_];
    __shared__ float sC[N_][CS_];
    int chunk = blockIdx.x, b = blockIdx.y, br = blockIdx.z;
    int d = threadIdx.x;
    int bbr = b * 2 + br;
    const float* dbl = br ? dblb : dblf;
    {
        int n = threadIdx.x >> 4, t4 = (threadIdx.x & 15) * 4;
        const float* gb = dbl + ((size_t)b * 48 + R_ + n) * L_ + chunk * CS_ + t4;
        async_ld_lds_b128(gb,              &sB[n][t4]);
        async_ld_lds_b128(gb + N_ * L_,    &sC[n][t4]);
        wait_async0();
    }
    __syncthreads();
    const float* Alog = br ? A_b_log : A_log;
    float Adn[N_];
#pragma unroll
    for (int n = 0; n < N_; ++n) Adn[n] = -__expf(Alog[d * N_ + n]);
    float h[N_];
    size_t base = (((size_t)bbr * CH_ + chunk) * DI_ + d);
#pragma unroll
    for (int n = 0; n < N_; ++n) h[n] = hin[base * N_ + n];
    float Dd = (br ? Dbv : Dv)[d];
    const float* dT = deltaT + (((size_t)bbr) * L_ + (size_t)chunk * CS_) * DI_ + d;
    for (int tt = 0; tt < CS_; ++tt) {
        int t  = chunk * CS_ + tt;
        int tu = br ? (L_ - 1 - t) : t;
        __builtin_prefetch(dT + (size_t)(tt + 4) * DI_, 0, 1);
        float delta = dT[(size_t)tt * DI_];
        float u = xT[((size_t)b * L_ + tu) * DI_ + d];
        float du = delta * u;
        float y = 0.f;
#pragma unroll
        for (int n = 0; n < N_; ++n) {
            h[n] = h[n] * __expf(delta * Adn[n]) + du * sB[n][tt];
            y += h[n] * sC[n][tt];
        }
        y += Dd * u;
        float zz = zT[((size_t)b * L_ + t) * DI_ + d];       // gate at scan coord t (module quirk)
        float g = zz / (1.f + __expf(-zz));
        ybuf[(((size_t)bbr) * DI_ + d) * L_ + t] = y * g;
    }
}

// ---------------------------------------------------------------------------
// Final: LayerNorm over l (per b,d) for both branches, silu(z) gates,
// backward flip, sum. out (b,d,l) f32.
// ---------------------------------------------------------------------------
__device__ __forceinline__ float block_reduce_sum(float v, float* sh)
{
    int tid = threadIdx.x;
    sh[tid] = v; __syncthreads();
    for (int s = 128; s > 0; s >>= 1) {
        if (tid < s) sh[tid] += sh[tid + s];
        __syncthreads();
    }
    float r = sh[0]; __syncthreads();
    return r;
}

__global__ void k_final(const float* __restrict__ ybuf, const float* __restrict__ xz,
                        const float* __restrict__ lnw, const float* __restrict__ lnb,
                        const float* __restrict__ ln1w, const float* __restrict__ ln1b,
                        float* __restrict__ out)
{
    __shared__ float sh[256];
    int d = blockIdx.x, b = blockIdx.y, tid = threadIdx.x;
    const float* yf = ybuf + (((size_t)b * 2 + 0) * DI_ + d) * L_;
    const float* yb = ybuf + (((size_t)b * 2 + 1) * DI_ + d) * L_;
    const float* z  = xz + ((size_t)b * E2_ + DI_ + d) * L_;
    float sf = 0.f, qf = 0.f, sb = 0.f, qb = 0.f;
    for (int l = tid; l < L_; l += 256) {
        float a = yf[l]; sf += a; qf += a * a;
        float c = yb[l]; sb += c; qb += c * c;
    }
    sf = block_reduce_sum(sf, sh); qf = block_reduce_sum(qf, sh);
    sb = block_reduce_sum(sb, sh); qb = block_reduce_sum(qb, sh);
    const float inv = 1.f / (float)L_;
    float mf = sf * inv, vf = qf * inv - mf * mf;
    float mb = sb * inv, vb = qb * inv - mb * mb;
    float rf = rsqrtf(vf + 1e-5f), rb = rsqrtf(vb + 1e-5f);
    for (int l = tid; l < L_; l += 256) {
        int lr = L_ - 1 - l;
        float z0 = z[l],  g0 = z0 / (1.f + __expf(-z0));
        float z1 = z[lr], g1 = z1 / (1.f + __expf(-z1));
        float o = ((yf[l]  - mf) * rf * lnw[l]   + lnb[l])  * g0
                + ((yb[lr] - mb) * rb * ln1w[lr] + ln1b[lr]) * g1;
        out[((size_t)b * DI_ + d) * L_ + l] = o;
    }
}

// ---------------------------------------------------------------------------
// Host launcher
// ---------------------------------------------------------------------------
extern "C" void kernel_launch(void* const* d_in, const int* in_sizes, int n_in,
                              void* d_out, int out_size, void* d_ws, size_t ws_size,
                              hipStream_t stream)
{
    const float* u          = (const float*)d_in[0];
    const float* in_proj_w  = (const float*)d_in[1];
    const float* conv_w     = (const float*)d_in[2];
    const float* conv_b     = (const float*)d_in[3];
    const float* x_proj_w   = (const float*)d_in[4];
    const float* dt_proj_w  = (const float*)d_in[5];
    const float* dt_bias    = (const float*)d_in[6];
    const float* A_log      = (const float*)d_in[7];
    const float* Dv         = (const float*)d_in[8];
    const float* x_proj_b_w = (const float*)d_in[9];
    const float* dt_proj_b_w= (const float*)d_in[10];
    const float* A_b_log    = (const float*)d_in[11];
    const float* Dbv        = (const float*)d_in[12];
    const float* ln_w       = (const float*)d_in[13];
    const float* ln_b       = (const float*)d_in[14];
    const float* ln1_w      = (const float*)d_in[15];
    const float* ln1_b      = (const float*)d_in[16];

    char* ws = (char*)d_ws;
    size_t off = 0;
    auto alloc = [&](size_t bytes) { size_t o = off; off += (bytes + 255) & ~(size_t)255; return o; };

    float* f_xz   = (float*)(ws + alloc((size_t)B_ * E2_ * L_ * 4));        // 33.5 MB
    bf16*  bf_u   = (bf16*) (ws + alloc((size_t)B_ * L_ * DI_ * 2));        //  8.4 MB
    bf16*  bf_w   = (bf16*) (ws + alloc((size_t)E2_ * DI_ * 2));
    bf16*  bf_xw  = (bf16*) (ws + alloc((size_t)48 * DI_ * 2));
    bf16*  bf_xwb = (bf16*) (ws + alloc((size_t)48 * DI_ * 2));
    float* f_xT   = (float*)(ws + alloc((size_t)B_ * L_ * DI_ * 4));        // 16.8 MB
    bf16*  bf_x   = (bf16*) (ws + alloc((size_t)B_ * L_ * DI_ * 2));        //  8.4 MB
    float* f_zT   = (float*)(ws + alloc((size_t)B_ * L_ * DI_ * 4));        // 16.8 MB
    float* f_dblf = (float*)(ws + alloc((size_t)B_ * 48 * L_ * 4));         //  3.1 MB
    float* f_dblb = (float*)(ws + alloc((size_t)B_ * 48 * L_ * 4));
    float* f_dlt  = (float*)(ws + alloc((size_t)B_ * 2 * L_ * DI_ * 4));    // 33.5 MB
    float* f_hfin = (float*)(ws + alloc((size_t)B_ * 2 * CH_ * DI_ * N_ * 4));
    float* f_dsum = (float*)(ws + alloc((size_t)B_ * 2 * CH_ * DI_ * 4));
    float* f_hin  = (float*)(ws + alloc((size_t)B_ * 2 * CH_ * DI_ * N_ * 4));
    float* f_y    = (float*)(ws + alloc((size_t)B_ * 2 * DI_ * L_ * 4));    // 33.5 MB

    const size_t NE = (size_t)B_ * DI_ * L_;   // 4,194,304

    // 1) casts
    k_cast <<<dim3((E2_ * DI_ + 255) / 256), 256, 0, stream>>>(in_proj_w,  bf_w,  E2_ * DI_);
    k_cast <<<dim3((48 * DI_ + 255) / 256),  256, 0, stream>>>(x_proj_w,   bf_xw,  48 * DI_);
    k_cast <<<dim3((48 * DI_ + 255) / 256),  256, 0, stream>>>(x_proj_b_w, bf_xwb, 48 * DI_);
    k_castu<<<dim3((unsigned)(NE / 256)),    256, 0, stream>>>(u, bf_u);

    // 2) in_proj GEMM (WMMA bf16)
    k_inproj<<<dim3(L_ / 16, E2_ / 16, B_), 32, 0, stream>>>(bf_w, bf_u, f_xz);

    // 3) depthwise conv + transpose + z extraction
    k_conv<<<dim3((unsigned)(NE / 256)), 256, 0, stream>>>(f_xz, conv_w, conv_b, f_xT, bf_x, f_zT);

    // 4) x_proj GEMMs, both branches (WMMA bf16)
    k_xproj<<<dim3(L_ / 16, 3, B_ * 2), 32, 0, stream>>>(bf_xw, bf_xwb, bf_x, f_dblf, f_dblb);

    // 5) dt_proj + softplus
    k_dt<<<dim3((unsigned)(2 * NE / 256)), 256, 0, stream>>>(f_dblf, f_dblb, dt_proj_w,
                                                             dt_proj_b_w, dt_bias, f_dlt);

    // 6-8) chunked parallel selective scan (async LDS staging)
    k_scan_local<<<dim3(CH_, B_, 2), 256, 0, stream>>>(f_dlt, f_xT, f_dblf, f_dblb,
                                                       A_log, A_b_log, f_hfin, f_dsum);
    k_scan_carry<<<dim3((B_ * 2 * DI_ * N_ + 255) / 256), 256, 0, stream>>>(f_hfin, f_dsum,
                                                                            A_log, A_b_log, f_hin);
    k_scan_emit<<<dim3(CH_, B_, 2), 256, 0, stream>>>(f_dlt, f_xT, f_zT, f_dblf, f_dblb,
                                                      A_log, A_b_log, Dv, Dbv, f_hin, f_y);

    // 9) layernorms + gates + flip + sum
    k_final<<<dim3(DI_, B_), 256, 0, stream>>>(f_y, f_xz, ln_w, ln_b, ln1_w, ln1_b, (float*)d_out);
}
